// Network_4982162063622
// MI455X (gfx1250) — matrix operand order repaired and checked by reading, not compiled, not run on hardware
//
#include <hip/hip_runtime.h>

typedef __attribute__((ext_vector_type(16))) _Float16 v16h;
typedef __attribute__((ext_vector_type(8)))  float    v8f;

#define BB   64
#define TT   800
#define UU   64
#define NNA  64
#define KKM  10
#define HH   512
#define OO   121
#define G4H  2048
#define KP0  608          // padded K for [x(3)|w(64)|h(512)] and [h|w|x] inputs
#define KT0  19           // KP0/32
#define KTH  16           // HH/32
#define HALLW 1536
#define ROWS (TT*BB)

#define WMMA(a,b,c) __builtin_amdgcn_wmma_f32_16x16x32_f16(false,(a),false,(b),(short)0,(c),false,false)

__device__ __forceinline__ float sigf(float x) { return 1.0f / (1.0f + expf(-x)); }

// A fragment (16x32 f16) gathered from an LDS row-major tile.
// Per ISA: lane L holds row M=L&15; VGPR q holds K = 16*(q>>2) + 8*(L>=16) + 2*(q&3) (+0,+1).
__device__ __forceinline__ v16h load_a_frag_lds(const _Float16* sA, int Kpad, int k0, int lane) {
  union { v16h v; unsigned u[8]; } f;
  const int row = lane & 15;
  const int hi8 = (lane >> 4) << 3;
  const _Float16* p = sA + row * Kpad + k0;
#pragma unroll
  for (int q = 0; q < 8; ++q) {
    int k = ((q >> 2) << 4) + hi8 + ((q & 3) << 1);
    f.u[q] = *(const unsigned*)(p + k);
  }
  return f.v;
}

// B fragment from pre-packed weights: 32 bytes contiguous per lane.
__device__ __forceinline__ v16h load_b_frag(const _Float16* Bp, long tileIdx, int lane) {
  return *(const v16h*)(Bp + ((tileIdx * 32 + (long)lane) << 4));
}

// CDNA5 async cache->LDS tile stage (GLOBAL_LOAD_ASYNC_TO_LDS_B128, ASYNCcnt).
// GVS mode: memaddr = SADDR(64) + VADDR(32) ; LDS addr = VDST. No VGPR data path.
// gSrc and ldsDst must be 16B aligned; bytes a multiple of 16.
__device__ __forceinline__ void async_stage_to_lds(void* ldsDst, const void* gSrc,
                                                   int bytes, int tid, int nthreads) {
  const unsigned ldsBase = (unsigned)(unsigned long long)ldsDst;  // low 32 bits = LDS offset
  for (int ofs = tid * 16; ofs < bytes; ofs += nthreads * 16) {
    unsigned lds = ldsBase + (unsigned)ofs;
    asm volatile("global_load_async_to_lds_b128 %0, %1, %2"
                 :: "v"(lds), "v"(ofs), "s"(gSrc)
                 : "memory");
  }
  asm volatile("s_wait_asynccnt 0x0" ::: "memory");
}

// ---------------- weight prep ----------------

// Pack f32 weight (Ksrc x Nsrc) into WMMA B-fragment order, zero-padded to (KT*32 x NT*16).
// B layout: lane half h element e holds K = 8h + (e&7) + 16*(e>>3); N = lane&15.
__global__ void pack_weight(const float* __restrict__ src, int Ksrc, int Nsrc,
                            _Float16* __restrict__ dst, int KT, int NT) {
  const int lane = threadIdx.x;          // 32 threads = one wave
  const int bidx = blockIdx.x;           // KT*NT blocks
  const int kt = bidx % KT, nt = bidx / KT;
  const int nn = nt * 16 + (lane & 15);
  const int kbase = kt * 32 + ((lane >> 4) << 3);
  _Float16* d = dst + (((long)(nt * KT + kt) * 32 + lane) << 4);
#pragma unroll
  for (int e = 0; e < 16; ++e) {
    int kk = kbase + (e & 7) + ((e >> 3) << 4);
    float v = (kk < Ksrc && nn < Nsrc) ? src[(long)kk * Nsrc + nn] : 0.0f;
    d[e] = (_Float16)v;
  }
}

// Combined layer-0 weight: rows 0..66 = W0 ([x|w]), rows 67..578 = R0 (h), pad to 608.
__global__ void build_wc0(const float* __restrict__ W0, const float* __restrict__ R0,
                          float* __restrict__ dst) {
  long idx = (long)blockIdx.x * 256 + threadIdx.x;
  if (idx >= (long)KP0 * G4H) return;
  int k = (int)(idx / G4H), n = (int)(idx % G4H);
  float v = 0.0f;
  if (k < 67)       v = W0[k * G4H + n];
  else if (k < 579) v = R0[(k - 67) * G4H + n];
  dst[idx] = v;
}

__global__ void zero32(unsigned* p, long n) {
  long i = (long)blockIdx.x * 256 + threadIdx.x;
  if (i < n) p[i] = 0u;
}

// Fill static x-slot (cols 576..578) + pad (579..607) of X1 rows, time-major row = t*64+b.
__global__ void x1_static(const float* __restrict__ strokes, _Float16* __restrict__ x1h) {
  long row = (long)blockIdx.x * 256 + threadIdx.x;
  if (row >= ROWS) return;
  long t = row >> 6, b = row & 63;
  _Float16* p = x1h + row * KP0;
  for (int k = 0; k < 3; ++k) p[576 + k] = (_Float16)strokes[b * (TT * 3) + t * 3 + k];
  for (int k = 579; k < KP0; ++k) p[k] = (_Float16)0.0f;
}

// ---------------- time-parallel GEMMs ----------------

// C(ROWS x 2048 f16) = A(ROWS x Kpad f16) @ Bpacked + bias. grid=(ROWS/16, 16), 8 waves/block.
__global__ void __launch_bounds__(256) gemm_xz(
    const _Float16* __restrict__ A, int Kpad, int KT,
    const _Float16* __restrict__ Bp, const float* __restrict__ bias,
    _Float16* __restrict__ C) {
  extern __shared__ _Float16 sA[];
  const int tid = threadIdx.x;
  const long m0 = blockIdx.x;
  async_stage_to_lds(sA, A + m0 * 16 * Kpad, 16 * Kpad * 2, tid, 256);
  __syncthreads();
  const int wave = tid >> 5, lane = tid & 31;
  const int nt = blockIdx.y * 8 + wave;
  v8f acc = {};
  for (int kt = 0; kt < KT; ++kt) {
    v16h a = load_a_frag_lds(sA, Kpad, kt * 32, lane);
    v16h b = load_b_frag(Bp, (long)nt * KT + kt, lane);
    acc = WMMA(a, b, acc);
  }
  const int col = (nt << 4) + (lane & 15);
  const int hi = (lane >> 4) << 3;
  const float bv = bias[col];
#pragma unroll
  for (int r = 0; r < 8; ++r) {
    long grow = m0 * 16 + r + hi;
    C[grow * G4H + col] = (_Float16)(acc[r] + bv);
  }
}

// Output projection: hall(ROWS x 1536) @ Wo(1536 x 121 pad 128) + bo, scatter to (B,T,O).
__global__ void __launch_bounds__(256) out_gemm(
    const _Float16* __restrict__ hall, const _Float16* __restrict__ Wop,
    const float* __restrict__ bo, float* __restrict__ outp) {
  __shared__ __align__(16) _Float16 sA[16 * HALLW];
  const int tid = threadIdx.x;
  const long m0 = blockIdx.x;
  async_stage_to_lds(sA, hall + m0 * 16 * HALLW, 16 * HALLW * 2, tid, 256);
  __syncthreads();
  const int wave = tid >> 5, lane = tid & 31;
  const int nt = wave;  // 8 n-tiles cover 128 padded cols
  v8f acc = {};
  for (int kt = 0; kt < 48; ++kt) {
    v16h a = load_a_frag_lds(sA, HALLW, kt * 32, lane);
    v16h b = load_b_frag(Wop, (long)nt * 48 + kt, lane);
    acc = WMMA(a, b, acc);
  }
  const int col = (nt << 4) + (lane & 15);
  if (col < OO) {
    const int hi = (lane >> 4) << 3;
    const float bv = bo[col];
#pragma unroll
    for (int r = 0; r < 8; ++r) {
      long gr = m0 * 16 + r + hi;
      long tt = gr >> 6, bb = gr & 63;
      outp[(bb * TT + tt) * OO + col] = acc[r] + bv;
    }
  }
}

// ---------------- fused sequential steps ----------------

// Layer 1/2 step: z = xz_t + h_{t-1}@R ; LSTM pointwise; h out to hall (+ X1 h-slot for layer1).
// Gate cols j, j+512, j+1024, j+1536 are N-tiles nb, nb+32, nb+64, nb+96 (same wave mod 8),
// so each wave keeps all 4 gate accumulators in registers and does the pointwise locally.
__global__ void __launch_bounds__(256) lstm_step(
    const _Float16* __restrict__ Rp, const _Float16* __restrict__ xz,
    _Float16* __restrict__ hbuf, float* __restrict__ cbuf,
    _Float16* __restrict__ hall, int hallOff,
    _Float16* __restrict__ x1h, int t) {
  __shared__ __align__(16) _Float16 sA[16 * HH];
  const int tid = threadIdx.x;
  const int rowBase = blockIdx.x * 16;  // 4 blocks x 16 batch rows
  async_stage_to_lds(sA, hbuf + (long)rowBase * HH, 16 * HH * 2, tid, 256);
  __syncthreads();
  const int wave = tid >> 5, lane = tid & 31;
#pragma unroll 1
  for (int i = 0; i < 4; ++i) {
    const int nb = wave + (i << 3);  // i-gate n-tile, 0..31
    v8f aI = {}, aF = {}, aG = {}, aO = {};
#pragma unroll 1
    for (int kt = 0; kt < KTH; ++kt) {
      v16h a  = load_a_frag_lds(sA, HH, kt * 32, lane);
      v16h bi = load_b_frag(Rp, (long)(nb      ) * KTH + kt, lane);
      v16h bf = load_b_frag(Rp, (long)(nb + 32 ) * KTH + kt, lane);
      v16h bg = load_b_frag(Rp, (long)(nb + 64 ) * KTH + kt, lane);
      v16h bo = load_b_frag(Rp, (long)(nb + 96 ) * KTH + kt, lane);
      aI = WMMA(a, bi, aI); aF = WMMA(a, bf, aF);
      aG = WMMA(a, bg, aG); aO = WMMA(a, bo, aO);
    }
    const int col = (nb << 4) + (lane & 15);  // 0..511
    const int hi = (lane >> 4) << 3;
#pragma unroll
    for (int r = 0; r < 8; ++r) {
      const int rr = r + hi, gb = rowBase + rr;
      const long orow = (long)t * BB + gb;
      const _Float16* xzr = xz + orow * G4H;
      float zi = aI[r] + (float)xzr[col];
      float zf = aF[r] + (float)xzr[col + 512];
      float zg = aG[r] + (float)xzr[col + 1024];
      float zo = aO[r] + (float)xzr[col + 1536];
      float cOld = cbuf[gb * HH + col];
      float c2 = sigf(zf) * cOld + sigf(zi) * tanhf(zg);
      float h2 = sigf(zo) * tanhf(c2);
      cbuf[gb * HH + col] = c2;
      _Float16 hh = (_Float16)h2;
      hbuf[gb * HH + col] = hh;
      hall[orow * HALLW + hallOff + col] = hh;
      if (x1h) x1h[orow * KP0 + col] = hh;
    }
  }
}

// Layer-0 step: z = [x|w|h]@Wc0 + b0 ; LSTM ; attention window -> w2, att, kappa.
__global__ void __launch_bounds__(256) layer0_step(
    const float* __restrict__ strokes, const float* __restrict__ trans,
    const _Float16* __restrict__ Wc0p, const float* __restrict__ b0,
    const float* __restrict__ Wd, const float* __restrict__ bd,
    _Float16* __restrict__ h0buf, float* __restrict__ c0buf,
    float* __restrict__ wbuf, float* __restrict__ kapbuf,
    _Float16* __restrict__ x1h, _Float16* __restrict__ hall,
    float* __restrict__ attOut, int t) {
  __shared__ __align__(16) _Float16 sU[16 * KP0];
  __shared__ float sH[16 * HH];
  __shared__ float sY[16 * 32];
  __shared__ float sKap[16 * KKM];
  __shared__ float sPhi[16 * 66];

  const int tid = threadIdx.x;
  const int rowBase = blockIdx.x * 16;

  // stage u = [x(3) | w(64) | h(512) | pad] as f16 (mixed-source gather)
  for (int idx = tid; idx < 16 * KP0; idx += 256) {
    int r = idx / KP0, k = idx - r * KP0;
    int gb = rowBase + r;
    float v;
    if (k < 3)        v = strokes[gb * (TT * 3) + t * 3 + k];
    else if (k < 67)  v = wbuf[gb * NNA + (k - 3)];
    else if (k < 579) v = (float)h0buf[gb * HH + (k - 67)];
    else              v = 0.0f;
    sU[idx] = (_Float16)v;
  }
  __syncthreads();

  const int wave = tid >> 5, lane = tid & 31;
#pragma unroll 1
  for (int i = 0; i < 4; ++i) {
    const int nb = wave + (i << 3);
    v8f aI = {}, aF = {}, aG = {}, aO = {};
#pragma unroll 1
    for (int kt = 0; kt < KT0; ++kt) {
      v16h a  = load_a_frag_lds(sU, KP0, kt * 32, lane);
      v16h bi = load_b_frag(Wc0p, (long)(nb      ) * KT0 + kt, lane);
      v16h bf = load_b_frag(Wc0p, (long)(nb + 32 ) * KT0 + kt, lane);
      v16h bg = load_b_frag(Wc0p, (long)(nb + 64 ) * KT0 + kt, lane);
      v16h bo = load_b_frag(Wc0p, (long)(nb + 96 ) * KT0 + kt, lane);
      aI = WMMA(a, bi, aI); aF = WMMA(a, bf, aF);
      aG = WMMA(a, bg, aG); aO = WMMA(a, bo, aO);
    }
    const int col = (nb << 4) + (lane & 15);
    const int hi = (lane >> 4) << 3;
    const float bI = b0[col], bF = b0[col + 512], bG = b0[col + 1024], bO = b0[col + 1536];
#pragma unroll
    for (int r = 0; r < 8; ++r) {
      const int rr = r + hi, gb = rowBase + rr;
      float c2 = sigf(aF[r] + bF) * c0buf[gb * HH + col] + sigf(aI[r] + bI) * tanhf(aG[r] + bG);
      float h2 = sigf(aO[r] + bO) * tanhf(c2);
      c0buf[gb * HH + col] = c2;
      sH[rr * HH + col] = h2;
      _Float16 hh = (_Float16)h2;
      h0buf[gb * HH + col] = hh;
      const long orow = (long)t * BB + gb;
      hall[orow * HALLW + col] = hh;
      x1h[orow * KP0 + col] = hh;
    }
  }
  __syncthreads();

  // y = exp(h2 @ Wd + bd): 16 rows x 30 outputs
  for (int j = tid; j < 16 * 30; j += 256) {
    int r = j / 30, jj = j - r * 30;
    float dot = bd[jj];
    const float* hr = sH + r * HH;
    for (int p = 0; p < HH; ++p) dot += hr[p] * Wd[p * 30 + jj];
    sY[r * 32 + jj] = expf(dot);
  }
  __syncthreads();
  {
    int r = tid >> 4, s = tid & 15;
    int gb = rowBase + r;
    if (s < KKM) {
      float k2 = kapbuf[gb * KKM + s] + sY[r * 32 + 20 + s];
      sKap[r * KKM + s] = k2;
      kapbuf[gb * KKM + s] = k2;
    }
  }
  __syncthreads();
  {
    int r = tid >> 4, s = tid & 15;
    for (int u = s; u < 65; u += 16) {
      float ph = 0.0f, fu = (float)(u + 1);
#pragma unroll
      for (int k = 0; k < KKM; ++k) {
        float d = sKap[r * KKM + k] - fu;
        ph += sY[r * 32 + k] * expf(-sY[r * 32 + 10 + k] * d * d);
      }
      sPhi[r * 66 + u] = ph;
    }
  }
  __syncthreads();
  // w2 = phi[:,:U] @ transcriptions[b]
  for (int j = tid; j < 16 * NNA; j += 256) {
    int r = j >> 6, n = j & 63;
    int gb = rowBase + r;
    float s2 = 0.0f;
    for (int u = 0; u < UU; ++u) s2 += sPhi[r * 66 + u] * trans[((long)gb * UU + u) * NNA + n];
    wbuf[gb * NNA + n] = s2;
    x1h[((long)t * BB + gb) * KP0 + HH + n] = (_Float16)s2;
  }
  // att = argmax over the 65 window weights
  if (tid < 16) {
    int r = tid, gb = rowBase + r;
    float best = sPhi[r * 66];
    int bi2 = 0;
    for (int u = 1; u < 65; ++u) {
      float p = sPhi[r * 66 + u];
      if (p > best) { best = p; bi2 = u; }
    }
    attOut[(long)gb * TT + t] = (float)bi2;
  }
}

// ---------------- host ----------------

extern "C" void kernel_launch(void* const* d_in, const int* in_sizes, int n_in,
                              void* d_out, int out_size, void* d_ws, size_t ws_size,
                              hipStream_t stream) {
  (void)in_sizes; (void)n_in; (void)out_size; (void)ws_size;
  const float* strokes = (const float*)d_in[0];
  const float* trans   = (const float*)d_in[1];
  const float* W0 = (const float*)d_in[2];
  const float* R0 = (const float*)d_in[3];
  const float* b0 = (const float*)d_in[4];
  const float* Wd = (const float*)d_in[5];
  const float* bd = (const float*)d_in[6];
  const float* W1 = (const float*)d_in[7];
  const float* R1 = (const float*)d_in[8];
  const float* b1 = (const float*)d_in[9];
  const float* W2 = (const float*)d_in[10];
  const float* R2 = (const float*)d_in[11];
  const float* b2 = (const float*)d_in[12];
  const float* Wo = (const float*)d_in[13];
  const float* bo = (const float*)d_in[14];

  char* ws = (char*)d_ws;
  auto carve = [&](size_t bytes) -> char* {
    char* p = ws;
    ws += (bytes + 255) & ~(size_t)255;
    return p;
  };
  float*    Wc0f  = (float*)   carve((size_t)KP0 * G4H * 4);
  _Float16* Wc0p  = (_Float16*)carve((size_t)KP0 * G4H * 2);
  _Float16* W1p   = (_Float16*)carve((size_t)KP0 * G4H * 2);
  _Float16* W2p   = (_Float16*)carve((size_t)KP0 * G4H * 2);
  _Float16* R1p   = (_Float16*)carve((size_t)HH * G4H * 2);
  _Float16* R2p   = (_Float16*)carve((size_t)HH * G4H * 2);
  _Float16* Wop   = (_Float16*)carve((size_t)HALLW * 128 * 2);
  _Float16* X1    = (_Float16*)carve((size_t)ROWS * KP0 * 2);
  _Float16* XZ    = (_Float16*)carve((size_t)ROWS * G4H * 2);
  _Float16* HALLb = (_Float16*)carve((size_t)ROWS * HALLW * 2);
  _Float16* h0buf = (_Float16*)carve((size_t)BB * HH * 2);
  float*    c0buf = (float*)   carve((size_t)BB * HH * 4);
  _Float16* hbuf  = (_Float16*)carve((size_t)BB * HH * 2);
  float*    cbuf  = (float*)   carve((size_t)BB * HH * 4);
  float*    wbuf  = (float*)   carve((size_t)BB * NNA * 4);
  float*    kapbuf= (float*)   carve((size_t)BB * KKM * 4);

  float* outp = (float*)d_out;
  float* attp = outp + (size_t)BB * TT * OO;

  // weight prep (fp32 -> f16 WMMA fragment packing)
  build_wc0<<<(KP0 * G4H + 255) / 256, 256, 0, stream>>>(W0, R0, Wc0f);
  pack_weight<<<KT0 * 128, 32, 0, stream>>>(Wc0f, KP0, G4H, Wc0p, KT0, 128);
  pack_weight<<<KT0 * 128, 32, 0, stream>>>(W1, 579, G4H, W1p, KT0, 128);
  pack_weight<<<KT0 * 128, 32, 0, stream>>>(W2, 579, G4H, W2p, KT0, 128);
  pack_weight<<<KTH * 128, 32, 0, stream>>>(R1, HH, G4H, R1p, KTH, 128);
  pack_weight<<<KTH * 128, 32, 0, stream>>>(R2, HH, G4H, R2p, KTH, 128);
  pack_weight<<<48 * 8, 32, 0, stream>>>(Wo, HALLW, OO, Wop, 48, 8);

  // static X1 slots + zero carries
  x1_static<<<(ROWS + 255) / 256, 256, 0, stream>>>(strokes, X1);
  zero32<<<(BB * HH / 2 + 255) / 256, 256, 0, stream>>>((unsigned*)h0buf, BB * HH / 2);
  zero32<<<(BB * HH + 255) / 256, 256, 0, stream>>>((unsigned*)c0buf, BB * HH);
  zero32<<<(BB * NNA + 255) / 256, 256, 0, stream>>>((unsigned*)wbuf, BB * NNA);
  zero32<<<(BB * KKM + 255) / 256, 256, 0, stream>>>((unsigned*)kapbuf, BB * KKM);

  // layer 0: sequential attention-LSTM scan
  for (int t = 0; t < TT; ++t)
    layer0_step<<<4, 256, 0, stream>>>(strokes, trans, Wc0p, b0, Wd, bd,
                                       h0buf, c0buf, wbuf, kapbuf, X1, HALLb, attp, t);

  // layer 1: batched input GEMM then sequential recurrence
  gemm_xz<<<dim3(ROWS / 16, 16), 256, 16 * KP0 * 2, stream>>>(X1, KP0, KT0, W1p, b1, XZ);
  zero32<<<(BB * HH / 2 + 255) / 256, 256, 0, stream>>>((unsigned*)hbuf, BB * HH / 2);
  zero32<<<(BB * HH + 255) / 256, 256, 0, stream>>>((unsigned*)cbuf, BB * HH);
  for (int t = 0; t < TT; ++t)
    lstm_step<<<4, 256, 0, stream>>>(R1p, XZ, hbuf, cbuf, HALLb, 512, X1, t);

  // layer 2 (X1 h-slot now holds h1)
  gemm_xz<<<dim3(ROWS / 16, 16), 256, 16 * KP0 * 2, stream>>>(X1, KP0, KT0, W2p, b2, XZ);
  zero32<<<(BB * HH / 2 + 255) / 256, 256, 0, stream>>>((unsigned*)hbuf, BB * HH / 2);
  zero32<<<(BB * HH + 255) / 256, 256, 0, stream>>>((unsigned*)cbuf, BB * HH);
  for (int t = 0; t < TT; ++t)
    lstm_step<<<4, 256, 0, stream>>>(R2p, XZ, hbuf, cbuf, HALLb, 1024, nullptr, t);

  // output projection
  out_gemm<<<ROWS / 16, 256, 0, stream>>>(HALLb, Wop, bo, outp);
}
